// GatedAttentionLayer_59322088293087
// MI455X (gfx1250) — matrix operand order repaired and checked by reading, not compiled, not run on hardware
//
#include <hip/hip_runtime.h>
#include <hip/hip_bf16.h>
#include <math.h>

#define H 16
#define HKV 8
#define DHEAD 128
#define ROT 32
#define CDIM 2048
#define SLEN 2048
#define CTXLEN 2048
#define NLAY 6
#define MQKV 6144  // 4096 (qg) + 1024 (k) + 1024 (v)

typedef __attribute__((ext_vector_type(16))) __bf16 bf16x16;
typedef __attribute__((ext_vector_type(8)))  __bf16 bf16x8;
typedef __attribute__((ext_vector_type(8)))  float  f32x8;
typedef __attribute__((ext_vector_type(4)))  int    i32x4;

// ---- CDNA5 async global->LDS path (ASYNCcnt-tracked DMA), with fallback ----
#ifdef __has_builtin
#  if __has_builtin(__builtin_amdgcn_global_load_async_to_lds_b128)
#    define HAVE_ASYNC_LDS 1
#  endif
#endif
#ifndef HAVE_ASYNC_LDS
#  define HAVE_ASYNC_LDS 0
#endif

#if HAVE_ASYNC_LDS
#define WAIT_ASYNC(n) asm volatile("s_wait_asynccnt " #n ::: "memory")
static __device__ __forceinline__ void async_copy16(const void* g, void* l) {
  __builtin_amdgcn_global_load_async_to_lds_b128(
      (__attribute__((address_space(1))) i32x4*)(g),
      (__attribute__((address_space(3))) i32x4*)(l), 0, 0);
}
#endif

static __device__ __forceinline__ __bf16 f2bf(float f) {
  unsigned u = __builtin_bit_cast(unsigned, f);
  unsigned r = u + 0x7FFFu + ((u >> 16) & 1u);
  return __builtin_bit_cast(__bf16, (unsigned short)(r >> 16));
}

static __device__ __forceinline__ bf16x16 cat8(bf16x8 lo, bf16x8 hp) {
  return __builtin_shufflevector(lo, hp, 0,1,2,3,4,5,6,7,8,9,10,11,12,13,14,15);
}

// ---------------------------------------------------------------- converts
__global__ __launch_bounds__(256) void convert_f32_bf16(
    const float* __restrict__ src, __bf16* __restrict__ dst, int n) {
  int idx = (blockIdx.x * 256 + threadIdx.x) * 4;
  if (idx < n) {
    float4 v = *(const float4*)(src + idx);
    union { __bf16 b[4]; uint2 u; } o;
    o.b[0] = f2bf(v.x); o.b[1] = f2bf(v.y); o.b[2] = f2bf(v.z); o.b[3] = f2bf(v.w);
    *(uint2*)(dst + idx) = o.u;
  }
}

// x: (C, S) f32 -> xT: (S, C) bf16   (token-major, K-contiguous B operand)
__global__ __launch_bounds__(256) void transpose_to_bf16(
    const float* __restrict__ x, __bf16* __restrict__ xT) {
  __shared__ float tile[32][33];
  int bc = blockIdx.x, br = blockIdx.y;
  int lr = threadIdx.x >> 5, lc = threadIdx.x & 31;
#pragma unroll
  for (int i = 0; i < 4; ++i)
    tile[lr + i * 8][lc] = x[(size_t)(br * 32 + lr + i * 8) * SLEN + bc * 32 + lc];
  __syncthreads();
#pragma unroll
  for (int i = 0; i < 4; ++i)
    xT[(size_t)(bc * 32 + lr + i * 8) * CDIM + br * 32 + lc] = f2bf(tile[lc][lr + i * 8]);
}

// ---------------------------------------------------------------- NT GEMM
// C[MxN] = A[MxK] @ BT[NxK]^T ; OUT_TRANS=1 writes CT[NxM] instead.
// 128x128x32 block tile, 8 waves of 64x32, double-buffered async LDS staging.
template <int OUT_TRANS>
__global__ __launch_bounds__(256) void gemm_nt_bf16(
    const __bf16* __restrict__ A, const __bf16* __restrict__ BT,
    float* __restrict__ C, int M, int N, int K) {
  constexpr int LDT = 40;  // 32 + 8 pad (80B rows -> conflict-free, 16B aligned)
  __shared__ __bf16 Al[2][128 * LDT];
  __shared__ __bf16 Bl[2][128 * LDT];
  int tid = threadIdx.x;
  int wid = tid >> 5, lane = tid & 31;
  int hi = lane >> 4, ln = lane & 15;
  int wm = (wid & 1) * 64;
  int wn = (wid >> 1) * 32;
  int m0 = blockIdx.y * 128;
  int n0 = blockIdx.x * 128;

  // this thread's two 16B staging chunks per operand tile
  int c0 = tid * 2, c1 = tid * 2 + 1;
  int r0 = c0 >> 2, o0 = (c0 & 3) * 8;
  int r1 = c1 >> 2, o1 = (c1 & 3) * 8;

  auto stage = [&](int buf, int k0) {
#if HAVE_ASYNC_LDS
    async_copy16(&A[(size_t)(m0 + r0) * K + k0 + o0], &Al[buf][r0 * LDT + o0]);
    async_copy16(&A[(size_t)(m0 + r1) * K + k0 + o1], &Al[buf][r1 * LDT + o1]);
    async_copy16(&BT[(size_t)(n0 + r0) * K + k0 + o0], &Bl[buf][r0 * LDT + o0]);
    async_copy16(&BT[(size_t)(n0 + r1) * K + k0 + o1], &Bl[buf][r1 * LDT + o1]);
#else
    *(uint4*)&Al[buf][r0 * LDT + o0] = *(const uint4*)&A[(size_t)(m0 + r0) * K + k0 + o0];
    *(uint4*)&Al[buf][r1 * LDT + o1] = *(const uint4*)&A[(size_t)(m0 + r1) * K + k0 + o1];
    *(uint4*)&Bl[buf][r0 * LDT + o0] = *(const uint4*)&BT[(size_t)(n0 + r0) * K + k0 + o0];
    *(uint4*)&Bl[buf][r1 * LDT + o1] = *(const uint4*)&BT[(size_t)(n0 + r1) * K + k0 + o1];
#endif
  };

  f32x8 acc[4][2] = {};
  int nk = K / 32;
  stage(0, 0);  // prologue: tile 0 in flight
  for (int t = 0; t < nk; ++t) {
    int buf = t & 1;
    if (t + 1 < nk) {
      stage(buf ^ 1, (t + 1) * 32);  // overlap next tile's DMA with compute
#if HAVE_ASYNC_LDS
      WAIT_ASYNC(4);                 // tile t's 4 ops retired; next 4 in flight
#endif
    } else {
#if HAVE_ASYNC_LDS
      WAIT_ASYNC(0);
#endif
    }
    __syncthreads();
    bf16x16 af[4], bfr[2];
#pragma unroll
    for (int mt = 0; mt < 4; ++mt) {
      const __bf16* p = &Al[buf][(wm + mt * 16 + ln) * LDT + hi * 8];
      af[mt] = cat8(*(const bf16x8*)p, *(const bf16x8*)(p + 16));
    }
#pragma unroll
    for (int nt = 0; nt < 2; ++nt) {
      const __bf16* p = &Bl[buf][(wn + nt * 16 + ln) * LDT + hi * 8];
      bfr[nt] = cat8(*(const bf16x8*)p, *(const bf16x8*)(p + 16));
    }
#pragma unroll
    for (int mt = 0; mt < 4; ++mt)
#pragma unroll
      for (int nt = 0; nt < 2; ++nt)
        acc[mt][nt] = __builtin_amdgcn_wmma_f32_16x16x32_bf16(
            false, af[mt], false, bfr[nt], (short)0, acc[mt][nt], false, false);
    __syncthreads();
  }
#pragma unroll
  for (int mt = 0; mt < 4; ++mt)
#pragma unroll
    for (int nt = 0; nt < 2; ++nt) {
      int mb = m0 + wm + mt * 16 + hi * 8;  // lane rows: mb..mb+7
      int n = n0 + wn + nt * 16 + ln;
      if (OUT_TRANS) {
        float* p = &C[(size_t)n * M + mb];
        float4 a, b;
        a.x = acc[mt][nt][0]; a.y = acc[mt][nt][1]; a.z = acc[mt][nt][2]; a.w = acc[mt][nt][3];
        b.x = acc[mt][nt][4]; b.y = acc[mt][nt][5]; b.z = acc[mt][nt][6]; b.w = acc[mt][nt][7];
        *(float4*)p = a;
        *(float4*)(p + 4) = b;
      } else {
#pragma unroll
        for (int v = 0; v < 8; ++v)
          C[(size_t)(mb + v) * N + n] = acc[mt][nt][v];
      }
    }
}

// ---------------------------------------------------------------- norm + rope
// One wave per (token, head-unit). Units: 0..15 q-heads, 16..23 k, 24..31 v.
__global__ __launch_bounds__(256) void norm_rope_kernel(
    const float* __restrict__ yT, const float* __restrict__ cosc,
    const float* __restrict__ sinc, const float* __restrict__ qnw,
    const float* __restrict__ knw, __bf16* __restrict__ qb,
    __bf16* __restrict__ kb, __bf16* __restrict__ vTb,
    float* __restrict__ cacheK, float* __restrict__ cacheV) {
  int wid = threadIdx.x >> 5, lane = threadIdx.x & 31;
  int g = blockIdx.x * 8 + wid;
  int s = g >> 5, u = g & 31;
  const float* row = yT + (size_t)s * MQKV;
  int mode, h, off;
  if (u < 16)      { mode = 0; h = u;      off = u * 256; }
  else if (u < 24) { mode = 1; h = u - 16; off = 4096 + h * DHEAD; }
  else             { mode = 2; h = u - 24; off = 5120 + h * DHEAD; }
  float x[4];
#pragma unroll
  for (int i = 0; i < 4; ++i) x[i] = row[off + lane + 32 * i];
  if (mode < 2) {
    float ss = 0.f;
#pragma unroll
    for (int i = 0; i < 4; ++i) ss += x[i] * x[i];
#pragma unroll
    for (int m = 1; m < 32; m <<= 1) ss += __shfl_xor(ss, m, 32);
    float r = rsqrtf(ss * (1.0f / 128.0f) + 1e-6f);
    const float* nw = (mode == 0) ? qnw : knw;
#pragma unroll
    for (int i = 0; i < 4; ++i) x[i] = x[i] * r * (1.0f + nw[lane + 32 * i]);
    // RoPE on dims 0..31: pair partner is lane^16 (same i==0 slot)
    float other = __shfl_xor(x[0], 16, 32);
    float cs = cosc[s * ROT + lane], sn = sinc[s * ROT + lane];
    float rot = (lane < 16) ? -other : other;
    x[0] = x[0] * cs + rot * sn;
  }
  if (mode == 0) {
    __bf16* q = qb + ((size_t)h * SLEN + s) * DHEAD;
#pragma unroll
    for (int i = 0; i < 4; ++i) q[lane + 32 * i] = f2bf(x[i]);
  } else if (mode == 1) {
    float* ck = cacheK + ((size_t)h * CTXLEN + s) * DHEAD;
    __bf16* kk = kb + ((size_t)h * SLEN + s) * DHEAD;
#pragma unroll
    for (int i = 0; i < 4; ++i) { int d = lane + 32 * i; ck[d] = x[i]; kk[d] = f2bf(x[i]); }
  } else {
    float* cv = cacheV + ((size_t)h * CTXLEN + s) * DHEAD;
    __bf16* vt = vTb + (size_t)h * DHEAD * CTXLEN;
#pragma unroll
    for (int i = 0; i < 4; ++i) { int d = lane + 32 * i; cv[d] = x[i]; vt[(size_t)d * CTXLEN + s] = f2bf(x[i]); }
  }
}

// ---------------------------------------------------------------- attention
// Flash attention: block = 4 waves x 16 query rows, 64-key chunks.
__global__ __launch_bounds__(128) void attn_kernel(
    const __bf16* __restrict__ qb, const __bf16* __restrict__ kb,
    const __bf16* __restrict__ vTb, const float* __restrict__ yT,
    __bf16* __restrict__ attnT) {
  constexpr int LK = 136, LV = 72, LP = 72;
  __shared__ __bf16 Kl[64 * LK];
  __shared__ __bf16 Vl[128 * LV];
  __shared__ __bf16 Pl[4 * 16 * LP];
  int h = blockIdx.y, kvh = h >> 1;
  int sblock = blockIdx.x * 64;
  int tid = threadIdx.x, wid = tid >> 5, lane = tid & 31;
  int hi = lane >> 4, ln = lane & 15;
  int sbase = sblock + wid * 16;
  // Q fragments for this wave's 16 rows: entire D=128 held in registers.
  const __bf16* qrow = qb + ((size_t)h * SLEN + (sbase + ln)) * DHEAD;
  bf16x16 qf[4];
#pragma unroll
  for (int kc = 0; kc < 4; ++kc) {
    int kb8 = kc * 32 + hi * 8;
    qf[kc] = cat8(*(const bf16x8*)&qrow[kb8], *(const bf16x8*)&qrow[kb8 + 16]);
  }
  f32x8 o[8] = {};
  float m[8], l[8];
#pragma unroll
  for (int v = 0; v < 8; ++v) { m[v] = -1e30f; l[v] = 0.f; }
  int nkt = (sblock >> 6) + 1;
  for (int kt = 0; kt < nkt; ++kt) {
    int t0 = kt * 64;
    __syncthreads();
#pragma unroll
    for (int i = 0; i < 8; ++i) {  // stage K (64x128) and V^T (128x64)
      int c = tid * 8 + i;
      int r = c >> 4, co = (c & 15) * 8;
      const __bf16* gK = &kb[((size_t)kvh * SLEN + t0 + r) * DHEAD + co];
      __bf16* lK = &Kl[r * LK + co];
      int r2 = c >> 3, co2 = (c & 7) * 8;
      const __bf16* gV = &vTb[((size_t)kvh * DHEAD + r2) * CTXLEN + t0 + co2];
      __bf16* lV = &Vl[r2 * LV + co2];
#if HAVE_ASYNC_LDS
      async_copy16(gK, lK);
      async_copy16(gV, lV);
#else
      *(uint4*)lK = *(const uint4*)gK;
      *(uint4*)lV = *(const uint4*)gV;
#endif
    }
#if HAVE_ASYNC_LDS
    WAIT_ASYNC(0);
#endif
    __syncthreads();
    // scores: 16 q-rows x 64 keys
    f32x8 st[4] = {};
#pragma unroll
    for (int nt = 0; nt < 4; ++nt)
#pragma unroll
      for (int kc = 0; kc < 4; ++kc) {
        const __bf16* p = &Kl[(nt * 16 + ln) * LK + kc * 32 + hi * 8];
        bf16x16 kf = cat8(*(const bf16x8*)p, *(const bf16x8*)(p + 16));
        st[nt] = __builtin_amdgcn_wmma_f32_16x16x32_bf16(
            false, qf[kc], false, kf, (short)0, st[nt], false, false);
      }
    // scale + causal mask
#pragma unroll
    for (int nt = 0; nt < 4; ++nt) {
      int tcol = t0 + nt * 16 + ln;
#pragma unroll
      for (int v = 0; v < 8; ++v) {
        int srow = sbase + hi * 8 + v;
        st[nt][v] = st[nt][v] * 0.08838834764831845f + (tcol > srow ? -1e9f : 0.0f);
      }
    }
    // online softmax (row stats via 16-lane shuffles)
#pragma unroll
    for (int v = 0; v < 8; ++v) {
      float mv = fmaxf(fmaxf(st[0][v], st[1][v]), fmaxf(st[2][v], st[3][v]));
#pragma unroll
      for (int msk = 1; msk < 16; msk <<= 1) mv = fmaxf(mv, __shfl_xor(mv, msk, 32));
      float mn = fmaxf(m[v], mv);
      float ef = __expf(m[v] - mn);
      m[v] = mn;
      l[v] *= ef;
#pragma unroll
      for (int dt = 0; dt < 8; ++dt) o[dt][v] *= ef;
#pragma unroll
      for (int nt = 0; nt < 4; ++nt) st[nt][v] = __expf(st[nt][v] - mn);
      float rs = st[0][v] + st[1][v] + st[2][v] + st[3][v];
#pragma unroll
      for (int msk = 1; msk < 16; msk <<= 1) rs += __shfl_xor(rs, msk, 32);
      l[v] += rs;
    }
    // transpose P (D-layout) -> A-layout through per-wave LDS scratch
    __bf16* pw = &Pl[wid * 16 * LP];
#pragma unroll
    for (int nt = 0; nt < 4; ++nt)
#pragma unroll
      for (int v = 0; v < 8; ++v)
        pw[(hi * 8 + v) * LP + nt * 16 + ln] = f2bf(st[nt][v]);
    bf16x16 pf[2];
#pragma unroll
    for (int pc = 0; pc < 2; ++pc) {
      const __bf16* p = &pw[ln * LP + pc * 32 + hi * 8];
      pf[pc] = cat8(*(const bf16x8*)p, *(const bf16x8*)(p + 16));
    }
    // O += P @ V
#pragma unroll
    for (int dt = 0; dt < 8; ++dt)
#pragma unroll
      for (int pc = 0; pc < 2; ++pc) {
        const __bf16* p = &Vl[(dt * 16 + ln) * LV + pc * 32 + hi * 8];
        bf16x16 vf = cat8(*(const bf16x8*)p, *(const bf16x8*)(p + 16));
        o[dt] = __builtin_amdgcn_wmma_f32_16x16x32_bf16(
            false, pf[pc], false, vf, (short)0, o[dt], false, false);
      }
  }
  // epilogue: normalize, gate, write token-major bf16
#pragma unroll
  for (int dt = 0; dt < 8; ++dt) {
    int d = dt * 16 + ln;
#pragma unroll
    for (int v = 0; v < 8; ++v) {
      int s = sbase + hi * 8 + v;
      float gate = yT[(size_t)s * MQKV + h * 256 + 128 + d];
      float val = (o[dt][v] / l[v]) * (1.0f / (1.0f + __expf(-gate)));
      attnT[(size_t)s * (H * DHEAD) + h * DHEAD + d] = f2bf(val);
    }
  }
}

// ---------------------------------------------------------------- launch
extern "C" void kernel_launch(void* const* d_in, const int* in_sizes, int n_in,
                              void* d_out, int out_size, void* d_ws, size_t ws_size,
                              hipStream_t stream) {
  const float* hidden = (const float*)d_in[0];
  const float* kv_in  = (const float*)d_in[1];
  const float* cosc   = (const float*)d_in[5];
  const float* sinc   = (const float*)d_in[6];
  const float* Wq     = (const float*)d_in[7];
  const float* Wk     = (const float*)d_in[8];
  const float* Wv     = (const float*)d_in[9];
  const float* Wo     = (const float*)d_in[10];
  const float* qnw    = (const float*)d_in[11];
  const float* knw    = (const float*)d_in[12];

  char* ws = (char*)d_ws;
  __bf16* Wcat  = (__bf16*)(ws);                 // 6144x2048 bf16 = 24 MB
  __bf16* Wob   = (__bf16*)(ws + 25165824);      //  8 MB
  __bf16* xT    = (__bf16*)(ws + 33554432);      //  8 MB
  float*  yT    = (float*) (ws + 41943040);      // 2048x6144 f32 = 48 MB
  __bf16* qb    = (__bf16*)(ws + 92274688);      //  8 MB
  __bf16* kbb   = (__bf16*)(ws + 100663296);     //  4 MB
  __bf16* vTb   = (__bf16*)(ws + 104857600);     //  4 MB
  __bf16* attnT = (__bf16*)(ws + 109051904);     //  8 MB

  float* out = (float*)d_out;
  float* kv_out = out + (size_t)CDIM * SLEN;
  (void)hipMemcpyAsync(kv_out, kv_in,
                       (size_t)2 * NLAY * HKV * CTXLEN * DHEAD * sizeof(float),
                       hipMemcpyDeviceToDevice, stream);

  convert_f32_bf16<<<(4096 * 2048) / 1024, 256, 0, stream>>>(Wq, Wcat, 4096 * 2048);
  convert_f32_bf16<<<(1024 * 2048) / 1024, 256, 0, stream>>>(Wk, Wcat + (size_t)4096 * 2048, 1024 * 2048);
  convert_f32_bf16<<<(1024 * 2048) / 1024, 256, 0, stream>>>(Wv, Wcat + (size_t)5120 * 2048, 1024 * 2048);
  convert_f32_bf16<<<(2048 * 2048) / 1024, 256, 0, stream>>>(Wo, Wob, 2048 * 2048);
  transpose_to_bf16<<<dim3(SLEN / 32, CDIM / 32), 256, 0, stream>>>(hidden, xT);

  // fused QKV projection: yT (S x 6144) token-major
  gemm_nt_bf16<1><<<dim3(SLEN / 128, MQKV / 128), 256, 0, stream>>>(
      Wcat, xT, yT, MQKV, SLEN, CDIM);

  norm_rope_kernel<<<(SLEN * 32) / 8, 256, 0, stream>>>(
      yT, cosc, sinc, qnw, knw, qb, kbb, vTb,
      kv_out, kv_out + (size_t)HKV * CTXLEN * DHEAD);

  attn_kernel<<<dim3(SLEN / 64, H), 128, 0, stream>>>(qb, kbb, vTb, yT, attnT);

  // output projection: (C,S) row-major straight into d_out
  gemm_nt_bf16<0><<<dim3(SLEN / 128, CDIM / 128), 256, 0, stream>>>(
      Wob, attnT, out, CDIM, SLEN, H * DHEAD);
}
